// SelfAttention_20040317403249
// MI455X (gfx1250) — matrix operand order repaired and checked by reading, not compiled, not run on hardware
//
#include <hip/hip_runtime.h>
#include <math.h>

#define B_ 8
#define S_ 2048
#define D_ 1024

typedef __attribute__((ext_vector_type(16))) __bf16 v16bf;
typedef __attribute__((ext_vector_type(8)))  __bf16 v8bf;
typedef __attribute__((ext_vector_type(4)))  __bf16 v4bf;
typedef __attribute__((ext_vector_type(2)))  __bf16 v2bf;
typedef __attribute__((ext_vector_type(8)))  float  v8f;

// Build a 16-element bf16 WMMA fragment from two 16B-aligned 8-element chunks.
__device__ __forceinline__ v16bf frag_from_2x8(const __bf16* lo_p, const __bf16* hi_p) {
  v8bf lo = *(const v8bf*)lo_p;
  v8bf hi = *(const v8bf*)hi_p;
  v16bf t;
#pragma unroll
  for (int e = 0; e < 8; ++e) { t[e] = lo[e]; t[e + 8] = hi[e]; }
  return t;
}

// ---------------------------------------------------------------------------
// QKV projection: out = X @ W + bias, f32 in, bf16 out (optionally transposed
// to [b][d][s] layout for V so the PV GEMM reads contiguous key runs).
// Block: 256 threads (8 waves). Tile: 128(M) x 128(N), K-step 32.
// Register-based double buffering: next tile's global loads issue before the
// WMMA block, conversion + LDS stores land after it (one barrier per K step).
// ---------------------------------------------------------------------------
#define PTM 128
#define PTN 128
#define PTK 32
#define PLDS 40   // padded row stride (elements); 80B rows keep 16B alignment

__global__ __launch_bounds__(256) void qkv_proj_kernel(
    const float* __restrict__ X, const float* __restrict__ W,
    const float* __restrict__ bias, __bf16* __restrict__ out,
    int transpose_out)
{
  __shared__ __bf16 Xs[2][PTM * PLDS];   // 2 x 10240 B
  __shared__ __bf16 Ws[2][PTN * PLDS];   // 2 x 10240 B

  const int tid  = threadIdx.x;
  const int wave = tid >> 5;
  const int lane = tid & 31;
  const int mbase = blockIdx.y * PTM;
  const int nbase = blockIdx.x * PTN;
  const int row = lane & 15;                  // A-frag row (per ISA layout)
  const int kba = (lane >> 4) * 8;            // A-frag K base
  const int col = lane & 15;                  // B/C-frag column
  const int kbb = (lane >> 4) * 8;            // B-frag K base

  float4 xv[4];                               // staged X quads (this thread)
  float  wv[8][2];                            // staged W k-pairs (this thread)

  auto load_tile = [&](int k0) {
#pragma unroll
    for (int it = 0; it < 4; ++it) {          // PTM*PTK/4/256 = 4
      int i = tid + it * 256;
      int r = i >> 3;                         // 8 quads per row
      int c = (i & 7) * 4;
      xv[it] = *(const float4*)(X + (size_t)(mbase + r) * D_ + k0 + c);
    }
#pragma unroll
    for (int it = 0; it < 8; ++it) {          // PTK*PTN/2/256 = 8
      int p = tid + it * 256;
      int n   = p & 127;
      int kk2 = (p >> 7) * 2;
      wv[it][0] = W[(size_t)(k0 + kk2)     * D_ + nbase + n];
      wv[it][1] = W[(size_t)(k0 + kk2 + 1) * D_ + nbase + n];
    }
  };
  auto store_tile = [&](int buf) {
#pragma unroll
    for (int it = 0; it < 4; ++it) {
      int i = tid + it * 256;
      int r = i >> 3;
      int c = (i & 7) * 4;
      v4bf o = { (__bf16)xv[it].x, (__bf16)xv[it].y, (__bf16)xv[it].z, (__bf16)xv[it].w };
      *(v4bf*)&Xs[buf][r * PLDS + c] = o;     // ds_store_b64
    }
#pragma unroll
    for (int it = 0; it < 8; ++it) {
      int p = tid + it * 256;
      int n   = p & 127;
      int kk2 = (p >> 7) * 2;
      v2bf o = { (__bf16)wv[it][0], (__bf16)wv[it][1] };
      *(v2bf*)&Ws[buf][n * PLDS + kk2] = o;   // ds_store_b32
    }
  };

  v8f acc[PTN / 16] = {};

  load_tile(0);
  store_tile(0);
  __syncthreads();

  for (int k0 = 0; k0 < D_; k0 += PTK) {
    const int cur = (k0 / PTK) & 1;
    const bool has_next = (k0 + PTK) < D_;
    if (has_next) load_tile(k0 + PTK);        // global loads overlap WMMAs below

    const __bf16* ap = &Xs[cur][(wave * 16 + row) * PLDS + kba];
    v16bf a = frag_from_2x8(ap, ap + 16);     // A: K = kba+0..7, kba+16..23
    v16bf bfr[PTN / 16];
#pragma unroll
    for (int f = 0; f < PTN / 16; ++f) {
      const __bf16* bp = &Ws[cur][(f * 16 + col) * PLDS + kbb];
      bfr[f] = frag_from_2x8(bp, bp + 16);    // B: K = kbb+0..7, kbb+16..23
    }
#pragma unroll
    for (int f = 0; f < PTN / 16; ++f)
      acc[f] = __builtin_amdgcn_wmma_f32_16x16x32_bf16(
          false, a, false, bfr[f], (short)0, acc[f], false, false);

    if (has_next) store_tile(cur ^ 1);        // convert + stash into other buffer
    __syncthreads();
  }

  // Epilogue: add bias, convert to bf16, store (optionally transposed).
#pragma unroll
  for (int f = 0; f < PTN / 16; ++f) {
    const int colg = nbase + f * 16 + col;
    const float bv = bias[colg];
#pragma unroll
    for (int e = 0; e < 8; ++e) {
      const int rowg = mbase + wave * 16 + e + ((lane >> 4) << 3);
      const __bf16 ob = (__bf16)(acc[f][e] + bv);
      if (!transpose_out) {
        out[(size_t)rowg * D_ + colg] = ob;             // [b*s][d]
      } else {
        const int bb = rowg >> 11;                       // / S_
        const int ss = rowg & (S_ - 1);
        out[((size_t)bb * D_ + colg) * S_ + ss] = ob;    // [b][d][s]
      }
    }
  }
}

// ---------------------------------------------------------------------------
// Flash attention: one block per (batch, 16-query tile). 8 waves.
// Per key tile (128 keys): wave w computes scores for keys [w*16, w*16+16),
// LDS online softmax, then each wave accumulates a 128-wide D slice of O.
// ---------------------------------------------------------------------------
#define KT   128
#define QSTR 1032          // 1024 + 8 pad (elements)
#define PSTR (KT + 8)

__global__ __launch_bounds__(256) void flash_attn_kernel(
    const __bf16* __restrict__ Qbf,
    const __bf16* __restrict__ Kbf,
    const __bf16* __restrict__ VT,   // [b][d][s]
    float* __restrict__ out)
{
  __shared__ __bf16 Qs[16 * QSTR];   // 33 KB
  __shared__ float  Ss[16 * KT];     // 8 KB
  __shared__ __bf16 Ps[16 * PSTR];   // 4.25 KB
  __shared__ float  red[16 * 16];
  __shared__ float  mrow[16], lrow[16], corr[16];

  const int tid  = threadIdx.x;
  const int wave = tid >> 5;
  const int lane = tid & 31;
  const int b     = blockIdx.y;
  const int qbase = blockIdx.x * 16;
  const int row = lane & 15;
  const int kba = (lane >> 4) * 8;
  const int col = lane & 15;
  const int kbb = (lane >> 4) * 8;
  const int dsl = wave * 128;                // this wave's output D slice

  // Stage the 16 x 1024 Q tile into LDS (bf16), b128 copies.
  const __bf16* Qg = Qbf + ((size_t)b * S_ + qbase) * D_;
#pragma unroll
  for (int it = 0; it < (16 * D_ / 8) / 256; ++it) {
    int i = tid + it * 256;
    int r = i >> 7;                          // 128 8-element chunks per row
    int c = (i & 127) * 8;
    *(v8bf*)&Qs[r * QSTR + c] = *(const v8bf*)(Qg + (size_t)r * D_ + c);
  }
  if (tid < 16) { mrow[tid] = -3.0e38f; lrow[tid] = 0.f; }
  v8f acc[8] = {};
  __syncthreads();

  const float scale = 0.03125f;              // 1/sqrt(1024)

  for (int kt = 0; kt < S_; kt += KT) {
    // --- 1. scores: this wave's 16 keys, full D reduction ---
    const __bf16* Kg = Kbf + ((size_t)b * S_ + kt + wave * 16) * D_;
    v8f sc = {};
    for (int d0 = 0; d0 < D_; d0 += 128) {   // 4 WMMA K-steps per iteration
      v16bf a4[4], b4[4];
#pragma unroll
      for (int u = 0; u < 4; ++u) {
        const int dd = d0 + u * 32;
        const __bf16* ap = &Qs[row * QSTR + dd + kba];
        a4[u] = frag_from_2x8(ap, ap + 16);
        const __bf16* bp = Kg + (size_t)col * D_ + dd + kbb;
        b4[u] = frag_from_2x8(bp, bp + 16);  // Kᵀ: d = kbb+0..7, kbb+16..23
        if (kt + KT < S_)                     // prefetch next key tile's lane data
          __builtin_prefetch(bp + (size_t)KT * D_, 0, 1);
      }
#pragma unroll
      for (int u = 0; u < 4; ++u)
        sc = __builtin_amdgcn_wmma_f32_16x16x32_bf16(
            false, a4[u], false, b4[u], (short)0, sc, false, false);
    }
#pragma unroll
    for (int e = 0; e < 8; ++e)
      Ss[(e + ((lane >> 4) << 3)) * KT + wave * 16 + col] = sc[e] * scale;
    __syncthreads();

    // --- 2. online softmax: 16 threads per query row ---
    const int r  = tid >> 4;
    const int c0 = (tid & 15) * 8;
    float sv[8];
    float mloc = -3.0e38f;
#pragma unroll
    for (int j = 0; j < 8; ++j) { sv[j] = Ss[r * KT + c0 + j]; mloc = fmaxf(mloc, sv[j]); }
    red[r * 16 + (tid & 15)] = mloc;
    __syncthreads();
    const float mold = mrow[r];
    float mnew = mold;
#pragma unroll
    for (int j = 0; j < 16; ++j) mnew = fmaxf(mnew, red[r * 16 + j]);
    const float alpha = __expf(mold - mnew);
    float psum = 0.f;
    v8bf pvv;
#pragma unroll
    for (int j = 0; j < 8; ++j) {
      float p = __expf(sv[j] - mnew);
      psum += p;
      pvv[j] = (__bf16)p;
    }
    *(v8bf*)&Ps[r * PSTR + c0] = pvv;         // one ds_store_b128
    __syncthreads();
    red[r * 16 + (tid & 15)] = psum;
    __syncthreads();
    if ((tid & 15) == 0) {
      float ls = 0.f;
#pragma unroll
      for (int j = 0; j < 16; ++j) ls += red[r * 16 + j];
      lrow[r] = lrow[r] * alpha + ls;
      mrow[r] = mnew;
      corr[r] = alpha;
    }
    __syncthreads();

    // --- 3. rescale accumulators, then P @ V for this wave's D slice ---
    float cf[8];
#pragma unroll
    for (int e = 0; e < 8; ++e) cf[e] = corr[e + ((lane >> 4) << 3)];
#pragma unroll
    for (int f = 0; f < 8; ++f)
#pragma unroll
      for (int e = 0; e < 8; ++e) acc[f][e] *= cf[e];

#pragma unroll
    for (int kk = 0; kk < KT; kk += 32) {
      const __bf16* ap = &Ps[row * PSTR + kk + kba];
      v16bf a = frag_from_2x8(ap, ap + 16);
      v16bf bv8[8];
#pragma unroll
      for (int f = 0; f < 8; ++f) {
        const int cold = dsl + f * 16 + col;
        const __bf16* bp =
            VT + ((size_t)b * D_ + cold) * S_ + kt + kk + kbb;   // contiguous keys
        bv8[f] = frag_from_2x8(bp, bp + 16);
        if (kt + KT < S_)
          __builtin_prefetch(bp + KT, 0, 1);  // next key tile, same d column
      }
#pragma unroll
      for (int f = 0; f < 8; ++f)
        acc[f] = __builtin_amdgcn_wmma_f32_16x16x32_bf16(
            false, a, false, bv8[f], (short)0, acc[f], false, false);
    }
    __syncthreads();
  }

  // Epilogue: O = acc / l, f32 output.
#pragma unroll
  for (int f = 0; f < 8; ++f) {
    const int cold = dsl + f * 16 + col;
#pragma unroll
    for (int e = 0; e < 8; ++e) {
      const int rowg = e + ((lane >> 4) << 3);
      out[((size_t)b * S_ + qbase + rowg) * D_ + cold] = acc[f][e] / lrow[rowg];
    }
  }
}

// ---------------------------------------------------------------------------
extern "C" void kernel_launch(void* const* d_in, const int* in_sizes, int n_in,
                              void* d_out, int out_size, void* d_ws, size_t ws_size,
                              hipStream_t stream) {
  (void)in_sizes; (void)n_in; (void)out_size; (void)ws_size;
  const float* in_k = (const float*)d_in[0];
  const float* in_q = (const float*)d_in[1];
  const float* in_v = (const float*)d_in[2];
  const float* Wk   = (const float*)d_in[3];
  const float* bk   = (const float*)d_in[4];
  const float* Wq   = (const float*)d_in[5];
  const float* bq   = (const float*)d_in[6];
  const float* Wv   = (const float*)d_in[7];
  const float* bv   = (const float*)d_in[8];

  const size_t N = (size_t)B_ * S_ * D_;
  __bf16* Kbf = (__bf16*)d_ws;                          // [b*s][d]
  __bf16* Qbf = Kbf + N;                                // [b*s][d]
  __bf16* VTb = Kbf + 2 * N;                            // [b][d][s]

  dim3 pgrid(D_ / PTN, (B_ * S_) / PTM);                // (8, 128)
  qkv_proj_kernel<<<pgrid, 256, 0, stream>>>(in_k, Wk, bk, Kbf, 0);
  qkv_proj_kernel<<<pgrid, 256, 0, stream>>>(in_q, Wq, bq, Qbf, 0);
  qkv_proj_kernel<<<pgrid, 256, 0, stream>>>(in_v, Wv, bv, VTb, 1);

  dim3 agrid(S_ / 16, B_);                              // (128, 8)
  flash_attn_kernel<<<agrid, 256, 0, stream>>>(Qbf, Kbf, VTb, (float*)d_out);
}
